// STMambaBlock_49838800503598
// MI455X (gfx1250) — matrix operand order repaired and checked by reading, not compiled, not run on hardware
//
#include <hip/hip_runtime.h>
#include <hip/hip_bf16.h>

// ---------------- problem dims (compile-time) ----------------
constexpr int Bn = 2, Tn = 4, Cn = 192, Hn = 32, Wn = 32;
constexpr int Ln  = Tn * Hn * Wn;   // 4096
constexpr int BL  = Bn * Ln;        // 8192
constexpr int DIN = 384, DSTATE = 16, DTRANK = 12, HID = 768;
constexpr int XDBL = 48;            // 44 (=12+16+16) padded to 48

typedef __attribute__((ext_vector_type(16))) __bf16 v16bf;
typedef __attribute__((ext_vector_type(8)))  float  v8f;

__device__ __forceinline__ __bf16 f2bf(float f) {
  union { float f; unsigned u; } c; c.f = f;
  unsigned r = c.u + 0x7FFFu + ((c.u >> 16) & 1u);   // round-to-nearest-even
  union { unsigned short s; __bf16 b; } o; o.s = (unsigned short)(r >> 16);
  return o.b;
}
__device__ __forceinline__ float siluf(float x) { return x / (1.f + __expf(-x)); }

// involutive sequence-index maps for the 4 scan directions
__device__ __forceinline__ int dir_map(int l, int dir) {
  if (dir & 1) l = Ln - 1 - l;                                  // flip along L
  if (dir & 2) { int t = l >> 10, r = l & 1023;                 // h <-> w swap
                 l = (t << 10) | ((r & 31) << 5) | (r >> 5); }
  return l;
}

// ---------------- layout / elementwise kernels ----------------
__global__ void k_permute_in(const float* __restrict__ xin,
                             float* __restrict__ x, float* __restrict__ acc) {
  int idx = blockIdx.x * blockDim.x + threadIdx.x;
  if (idx >= BL * Cn) return;
  int c = idx % Cn, row = idx / Cn, l = row % Ln, b = row / Ln;
  int t = l >> 10, hw = l & 1023;
  float v = xin[(((long)(b * Tn + t) * Cn + c)) * (Hn * Wn) + hw];
  x[idx] = v; acc[idx] = v;
}

__global__ void k_permute_out(const float* __restrict__ acc, float* __restrict__ out) {
  int idx = blockIdx.x * blockDim.x + threadIdx.x;
  if (idx >= BL * Cn) return;
  int hw = idx & 1023;
  int c  = (idx >> 10) % Cn;
  int t  = (idx / (1024 * Cn)) % Tn;
  int b  =  idx / (1024 * Cn * Tn);
  out[idx] = acc[((long)(b * Ln + (t << 10) + hw)) * Cn + c];
}

// rmsnorm writing both f32 and bf16 copies of the result
__global__ void k_rmsnorm(const float* __restrict__ src, const float* __restrict__ w,
                          float* __restrict__ dst, __bf16* __restrict__ dstb) {
  int row  = blockIdx.x * (blockDim.x >> 5) + (threadIdx.x >> 5);
  int lane = threadIdx.x & 31;
  if (row >= BL) return;
  const float* p = src + (long)row * Cn;
  float s = 0.f;
  for (int c = lane; c < Cn; c += 32) { float v = p[c]; s += v * v; }
  for (int off = 16; off; off >>= 1) s += __shfl_xor(s, off, 32);
  float inv = rsqrtf(s / (float)Cn + 1e-5f);
  for (int c = lane; c < Cn; c += 32) {
    float v = p[c] * inv * w[c];
    dst[(long)row * Cn + c]  = v;
    dstb[(long)row * Cn + c] = f2bf(v);
  }
}

// gather bf16 -> bf16 (pure permuted copy)
__global__ void k_gather(const __bf16* __restrict__ xnb, __bf16* __restrict__ seqb, int dir) {
  int idx = blockIdx.x * blockDim.x + threadIdx.x;
  if (idx >= BL * Cn) return;
  int c = idx % Cn, row = idx / Cn, l = row % Ln, b = row / Ln;
  seqb[idx] = xnb[((long)(b * Ln + dir_map(l, dir))) * Cn + c];
}

__global__ void k_acc_gather(const float* __restrict__ outd, float* __restrict__ acc, int dir) {
  int idx = blockIdx.x * blockDim.x + threadIdx.x;
  if (idx >= BL * Cn) return;
  int c = idx % Cn, row = idx / Cn, l = row % Ln, b = row / Ln;
  acc[idx] += outd[((long)(b * Ln + dir_map(l, dir))) * Cn + c];
}

__global__ void k_f32_to_bf16(const float* __restrict__ src, __bf16* __restrict__ dst, int n) {
  int i = blockIdx.x * blockDim.x + threadIdx.x;
  if (i < n) dst[i] = f2bf(src[i]);
}

__global__ void k_pad_xpw(const float* __restrict__ src, __bf16* __restrict__ dst) {
  int i = blockIdx.x * blockDim.x + threadIdx.x;
  if (i >= XDBL * DIN) return;
  int r = i / DIN;
  dst[i] = (r < DTRANK + 2 * DSTATE) ? f2bf(src[i]) : f2bf(0.f);
}

// causal depthwise conv1d (k=4) over L per batch + bias + SiLU; input = xz[:, :384]
__global__ void k_conv_silu(const float* __restrict__ xz, const float* __restrict__ cw,
                            const float* __restrict__ cb, float* __restrict__ xc,
                            __bf16* __restrict__ xcb) {
  int idx = blockIdx.x * blockDim.x + threadIdx.x;
  if (idx >= BL * DIN) return;
  int d = idx % DIN, row = idx / DIN, l = row % Ln, b = row / Ln;
  float s = cb[d];
#pragma unroll
  for (int k = 0; k < 4; ++k) {
    int ls = l - 3 + k;
    if (ls >= 0) s += cw[d * 4 + k] * xz[((long)(b * Ln + ls)) * HID + d];
  }
  float v = siluf(s);
  xc[idx]  = v;
  xcb[idx] = f2bf(v);
}

__global__ void k_dtproj(const float* __restrict__ xdbl, const float* __restrict__ dw,
                         const float* __restrict__ db, float* __restrict__ dt) {
  int idx = blockIdx.x * blockDim.x + threadIdx.x;
  if (idx >= BL * DIN) return;
  int d = idx % DIN, row = idx / DIN;
  float s = db[d];
#pragma unroll
  for (int j = 0; j < DTRANK; ++j) s += xdbl[(long)row * XDBL + j] * dw[d * DTRANK + j];
  dt[idx] = (s > 20.f) ? s : log1pf(__expf(s));   // softplus
}

// y_bf = y * silu(z); bf16 output feeds the out_proj GEMM
__global__ void k_gate(const float* __restrict__ xz, const float* __restrict__ y,
                       __bf16* __restrict__ yb) {
  int idx = blockIdx.x * blockDim.x + threadIdx.x;
  if (idx >= BL * DIN) return;
  int d = idx % DIN; long row = idx / DIN;
  yb[idx] = f2bf(y[idx] * siluf(xz[row * HID + DIN + d]));
}

__global__ void k_dwconv_gelu(const float* __restrict__ h1, const float* __restrict__ ww,
                              const float* __restrict__ wb, __bf16* __restrict__ h2b) {
  int idx = blockIdx.x * blockDim.x + threadIdx.x;
  if (idx >= BL * HID) return;
  int ch = idx % HID, row = idx / HID, l = row % Ln, b = row / Ln;
  int t = l >> 10, r = l & 1023, h = r >> 5, w = r & 31;
  float s = wb[ch];
#pragma unroll
  for (int dy = 0; dy < 3; ++dy)
#pragma unroll
    for (int dx = 0; dx < 3; ++dx) {
      int hh = h + dy - 1, wc = w + dx - 1;
      if (hh >= 0 && hh < Hn && wc >= 0 && wc < Wn)
        s += ww[(dy * 3 + dx) * HID + ch] *
             h1[((long)(b * Ln + (t << 10) + (hh << 5) + wc)) * HID + ch];
    }
  float g = 0.5f * s * (1.f + erff(s * 0.70710678118654752f));   // exact GELU
  h2b[idx] = f2bf(g);
}

// ---------------- selective scan: 1 thread per (b,d) channel ----------------
__global__ void __launch_bounds__(DIN) k_scan(const float* __restrict__ xc,
                                              const float* __restrict__ dt,
                                              const float* __restrict__ xdbl,
                                              const float* __restrict__ A_log,
                                              const float* __restrict__ Dv,
                                              float* __restrict__ y) {
  constexpr int CHUNK = 32;
  __shared__ float sB[CHUNK][DSTATE];
  __shared__ float sC[CHUNK][DSTATE];
  const int b = blockIdx.x, d = threadIdx.x;
  float a[DSTATE], h[DSTATE];
#pragma unroll
  for (int n = 0; n < DSTATE; ++n) { a[n] = -__expf(A_log[d * DSTATE + n]); h[n] = 0.f; }
  const float Dd = Dv[d];
  for (int l0 = 0; l0 < Ln; l0 += CHUNK) {
    __syncthreads();
    for (int i = d; i < CHUNK * DSTATE; i += DIN) {
      int s = i >> 4, n = i & 15;
      long row = (long)(b * Ln + l0 + s) * XDBL;
      sB[s][n] = xdbl[row + DTRANK + n];
      sC[s][n] = xdbl[row + DTRANK + DSTATE + n];
    }
    __syncthreads();
    for (int s = 0; s < CHUNK; ++s) {
      long ridx = (long)(b * Ln + l0 + s);
      float dtv = dt[ridx * DIN + d];
      float u   = xc[ridx * DIN + d];
      float du  = dtv * u;
      float out = 0.f;
#pragma unroll
      for (int n = 0; n < DSTATE; ++n) {
        float dA = __expf(dtv * a[n]);
        h[n] = dA * h[n] + du * sB[s][n];
        out += h[n] * sC[s][n];
      }
      y[ridx * DIN + d] = out + Dd * u;
    }
  }
}

// ------- bf16 WMMA GEMM: Out[M,N] = A[M,K] @ W[N,K]^T (+bias)(+=), bf16 operands -----
// Fully shape-specialized: wave tile = 16(M) x NT*16(N); no predicates in hot loop.
// block = 4 waves; grid = (M/64, N/(NT*16)) -- exact division by construction.
template <int K, int N, int NT, int FLAGS>
__global__ void __launch_bounds__(128) k_gemm(const __bf16* __restrict__ A,
                                              const __bf16* __restrict__ W,
                                              const float* __restrict__ bias,
                                              float* __restrict__ Out) {
  const int lane  = threadIdx.x;
  const int mtile = blockIdx.x * 4 + threadIdx.y;
  const int n0    = blockIdx.y * (NT * 16);
  const int m     = lane & 15;
  const int half  = lane >> 4;
  const long row  = (long)mtile * 16 + m;

  v8f acc[NT] = {};
  const __bf16* arow = A + row * K;
#pragma unroll
  for (int k0 = 0; k0 < K; k0 += 32) {
    const __bf16* ap = arow + k0 + half * 8;
    if (k0 + 32 < K) __builtin_prefetch(ap + 32, 0, 1);   // global_prefetch_b8
    union { v16bf v; float4 f[2]; } af;
    af.f[0] = *(const float4*)(ap);        // bf16 k0+8h    .. +7  -> elems 0..7
    af.f[1] = *(const float4*)(ap + 16);   // bf16 k0+16+8h .. +7  -> elems 8..15
#pragma unroll
    for (int nt = 0; nt < NT; ++nt) {
      const __bf16* wp = W + (long)(n0 + nt * 16 + m) * K + k0 + half * 8;
      union { v16bf v; float4 f[2]; } bfr;
      bfr.f[0] = *(const float4*)(wp);
      bfr.f[1] = *(const float4*)(wp + 16);
      acc[nt] = __builtin_amdgcn_wmma_f32_16x16x32_bf16(
          false, af.v, false, bfr.v, (short)0, acc[nt], false, false);
    }
  }
#pragma unroll
  for (int nt = 0; nt < NT; ++nt) {
    int ncol = n0 + nt * 16 + m;
#pragma unroll
    for (int r = 0; r < 8; ++r) {
      long orow = (long)mtile * 16 + half * 8 + r;   // D layout: M = r + 8*half
      float v = acc[nt][r];
      if (FLAGS & 1) v += bias[ncol];
      float* o = Out + orow * N + ncol;
      if (FLAGS & 2) v += *o;
      *o = v;
    }
  }
}

// ---------------- host-side orchestration ----------------
static inline int cdiv(int a, int b) { return (a + b - 1) / b; }

extern "C" void kernel_launch(void* const* d_in, const int* in_sizes, int n_in,
                              void* d_out, int out_size, void* d_ws, size_t ws_size,
                              hipStream_t stream) {
  const float* x_in      = (const float*)d_in[0];
  const float* norm1_w   = (const float*)d_in[1];
  const float* norm2_w   = (const float*)d_in[2];
  const float* in_proj_w = (const float*)d_in[3];
  const float* conv_w    = (const float*)d_in[4];
  const float* conv_b    = (const float*)d_in[5];
  const float* x_proj_w  = (const float*)d_in[6];
  const float* dt_proj_w = (const float*)d_in[7];
  const float* dt_proj_b = (const float*)d_in[8];
  const float* A_log     = (const float*)d_in[9];
  const float* Dv        = (const float*)d_in[10];
  const float* out_proj_w= (const float*)d_in[11];
  const float* fc1_w     = (const float*)d_in[12];
  const float* fc1_b     = (const float*)d_in[13];
  const float* dw_w      = (const float*)d_in[14];
  const float* dw_b      = (const float*)d_in[15];
  const float* fc2_w     = (const float*)d_in[16];
  const float* fc2_b     = (const float*)d_in[17];

  float* ws = (float*)d_ws;
  const size_t NC = (size_t)BL * Cn;

  // f32 regions
  float* x    = ws;
  float* xn   = ws + 1 * NC;
  float* acc  = ws + 2 * NC;
  float* outd = ws + 3 * NC;
  float* xz   = ws + 4 * NC;                       // BL x 768  (also MLP h1)
  float* xc   = xz + (size_t)BL * HID;             // BL x 384  (overlaid by h2_bf in MLP)
  float* dt   = xc + (size_t)BL * DIN;             // BL x 384  (overlaid by y_bf after scan)
  float* y    = dt + (size_t)BL * DIN;             // BL x 384
  float* xdbl = y  + (size_t)BL * DIN;             // BL x 48
  float* f32_end = xdbl + (size_t)BL * XDBL;

  // bf16 regions
  __bf16* xn_bf  = (__bf16*)f32_end;               // BL x 192 (xn / xn2)
  __bf16* seq_bf = xn_bf  + NC;                    // BL x 192
  __bf16* xc_bf  = seq_bf + NC;                    // BL x 384
  __bf16* w_in   = xc_bf  + (size_t)BL * DIN;      // 768 x 192
  __bf16* w_xp   = w_in   + (size_t)HID * Cn;      // 48 x 384 (padded)
  __bf16* w_out  = w_xp   + (size_t)XDBL * DIN;    // 192 x 384
  __bf16* w_fc1  = w_out  + (size_t)Cn * DIN;      // 768 x 192
  __bf16* w_fc2  = w_fc1  + (size_t)HID * Cn;      // 192 x 768
  // overlays of dead f32 regions (each fits within the region it overlays)
  __bf16* y_bf   = (__bf16*)dt;                    // BL x 384 bf16 over BL x 384 f32 /2
  __bf16* h2_bf  = (__bf16*)xc;                    // BL x 768 bf16 over BL x 384 f32

  const dim3 gblk(32, 4);
  const int TPB = 256;

  // one-time weight quantization (bf16)
  k_f32_to_bf16<<<cdiv(HID * Cn, TPB), TPB, 0, stream>>>(in_proj_w, w_in, HID * Cn);
  k_pad_xpw<<<cdiv(XDBL * DIN, TPB), TPB, 0, stream>>>(x_proj_w, w_xp);
  k_f32_to_bf16<<<cdiv(Cn * DIN, TPB), TPB, 0, stream>>>(out_proj_w, w_out, Cn * DIN);
  k_f32_to_bf16<<<cdiv(HID * Cn, TPB), TPB, 0, stream>>>(fc1_w, w_fc1, HID * Cn);
  k_f32_to_bf16<<<cdiv(Cn * HID, TPB), TPB, 0, stream>>>(fc2_w, w_fc2, Cn * HID);

  k_permute_in<<<cdiv(BL * Cn, TPB), TPB, 0, stream>>>(x_in, x, acc);
  k_rmsnorm<<<BL / 8, TPB, 0, stream>>>(x, norm1_w, xn, xn_bf);

  for (int dir = 0; dir < 4; ++dir) {
    k_gather<<<cdiv(BL * Cn, TPB), TPB, 0, stream>>>(xn_bf, seq_bf, dir);
    k_gemm<Cn, HID, 4, 0><<<dim3(BL / 64, HID / 64), gblk, 0, stream>>>(seq_bf, w_in, nullptr, xz);
    k_conv_silu<<<cdiv(BL * DIN, TPB), TPB, 0, stream>>>(xz, conv_w, conv_b, xc, xc_bf);
    k_gemm<DIN, XDBL, 3, 0><<<dim3(BL / 64, 1), gblk, 0, stream>>>(xc_bf, w_xp, nullptr, xdbl);
    k_dtproj<<<cdiv(BL * DIN, TPB), TPB, 0, stream>>>(xdbl, dt_proj_w, dt_proj_b, dt);
    k_scan<<<Bn, DIN, 0, stream>>>(xc, dt, xdbl, A_log, Dv, y);
    k_gate<<<cdiv(BL * DIN, TPB), TPB, 0, stream>>>(xz, y, y_bf);
    k_gemm<DIN, Cn, 4, 0><<<dim3(BL / 64, Cn / 64), gblk, 0, stream>>>(y_bf, w_out, nullptr, outd);
    k_acc_gather<<<cdiv(BL * Cn, TPB), TPB, 0, stream>>>(outd, acc, dir);
  }

  // MLP: rmsnorm -> fc1+bias -> depthwise 3x3 + GELU -> fc2+bias (accumulated into acc)
  k_rmsnorm<<<BL / 8, TPB, 0, stream>>>(acc, norm2_w, xn, xn_bf);
  k_gemm<Cn, HID, 4, 1><<<dim3(BL / 64, HID / 64), gblk, 0, stream>>>(xn_bf, w_fc1, fc1_b, xz);
  k_dwconv_gelu<<<cdiv(BL * HID, TPB), TPB, 0, stream>>>(xz, dw_w, dw_b, h2_bf);
  k_gemm<HID, Cn, 4, 3><<<dim3(BL / 64, Cn / 64), gblk, 0, stream>>>(h2_bf, w_fc2, fc2_b, acc);

  k_permute_out<<<cdiv(BL * Cn, TPB), TPB, 0, stream>>>(acc, (float*)d_out);
}
